// DGDAGRNN_72834055405595
// MI455X (gfx1250) — matrix operand order repaired and checked by reading, not compiled, not run on hardware
//
#include <hip/hip_runtime.h>
#include <hip/hip_bf16.h>
#include <math.h>

typedef __attribute__((ext_vector_type(16))) _Float16 v16h;
typedef __attribute__((ext_vector_type(8)))  float    v8f;

#define NLV   12
#define NPL   10000
#define NTOT  (NLV * NPL)
#define KFAN  3
#define EPL   (NPL * KFAN)
#define HDIM  100
#define FDIM  4
#define HID   50
#define DMID  30
#define KPAD  128      // K pad for H=100
#define NPAD1 64       // 50 -> 64
#define NPAD2 112      // 100 -> 112
#define GHPAD 304      // 300 -> 304
#define CLSP  32       // 30 -> 32
#define HARD_INV_T 100.0f

// ---------------- WMMA fragment helpers (wave32, 16x16x32 f16) ----------------
// A: M x K row-major (LDS), 16-bit A layout: lanes 0-15 -> K0..7,K16..23 pairs,
// lanes 16-31 -> K8..15,K24..31 pairs (ISA 7.12.2).
__device__ __forceinline__ v16h load_a_frag(const _Float16* A, int lda, int m0,
                                            int k0, int lane) {
  const _Float16* r = A + (m0 + (lane & 15)) * lda + k0 + ((lane >> 4) << 3);
  v16h f;
#pragma unroll
  for (int v = 0; v < 4; ++v) { f[2*v]   = r[2*v];      f[2*v+1] = r[2*v+1]; }
#pragma unroll
  for (int v = 0; v < 4; ++v) { f[8+2*v] = r[16+2*v];   f[9+2*v] = r[17+2*v]; }
  return f;
}

// B stored transposed: BT[n][k], row stride ldk. lanes 0-15: N=lane, K=k0+0..7,k0+16..23;
// lanes 16-31: N=lane-16, K=k0+8..15,k0+24..31 (same k order as A).
__device__ __forceinline__ v16h load_b_frag(const _Float16* BT, int ldk, int n0,
                                            int k0, int lane) {
  const _Float16* r = BT + (n0 + (lane & 15)) * ldk + k0 + ((lane >> 4) << 3);
  v16h f;
#pragma unroll
  for (int v = 0; v < 8; ++v) { f[v] = r[v]; f[8+v] = r[16+v]; }
  return f;
}

template <int KTOT>
__device__ __forceinline__ v8f mma_tile(const _Float16* A, int lda, int m0,
                                        const _Float16* BT, int ldk, int n0,
                                        int lane) {
  v8f c = {};
#pragma unroll
  for (int k0 = 0; k0 < KTOT; k0 += 32) {
    v16h a = load_a_frag(A, lda, m0, k0, lane);
    v16h b = load_b_frag(BT, ldk, n0, k0, lane);
    c = __builtin_amdgcn_wmma_f32_16x16x32_f16(false, a, false, b,
                                               (short)0, c, false, false);
  }
  return c;
}

// ---------------- weight prep: f32 -> padded f16 BT[n][k] ----------------
__global__ void prep_bt_kernel(const float* __restrict__ w, _Float16* __restrict__ bt,
                               int K, int N, int Kp, int Np, int transpose) {
  int idx = blockIdx.x * 256 + threadIdx.x;
  if (idx >= Kp * Np) return;
  int n = idx / Kp, k = idx - n * Kp;
  float v = 0.0f;
  if (n < N && k < K) v = transpose ? w[k * N + n] : w[n * K + k];
  bt[idx] = (_Float16)v;
}

// ---------------- edge pre-MLP (+ aggregate into agg) ----------------
// 48 edges / block (= 16 dst nodes in forward triple mode).
__global__ void __launch_bounds__(128) edge_pre_kernel(
    const float* __restrict__ state,
    const int* __restrict__ gidx,     // per-edge gather node id (level-offset applied)
    const int* __restrict__ sidx,     // nullptr -> forward triple mode; else scatter ids
    int scat_base,
    const _Float16* __restrict__ w1bt, const float* __restrict__ b1,
    const _Float16* __restrict__ w2bt, const float* __restrict__ b2,
    float* __restrict__ agg) {        // [NPL][NPAD2]
  __shared__ _Float16 lA[48 * KPAD];
  __shared__ _Float16 lH[48 * NPAD1];
  __shared__ float    lP[48 * NPAD2];
  __shared__ int      sg[48];
  __shared__ int      ss[48];
  const int tid = threadIdx.x, lane = tid & 31, wave = tid >> 5;
  const int ebase = blockIdx.x * 48;
  if (tid < 48) {
    sg[tid] = gidx[ebase + tid];
    ss[tid] = sidx ? (sidx[ebase + tid] - scat_base) : 0;
  }
  __syncthreads();
  for (int i = tid; i < 48 * KPAD; i += 128) {      // gather + f16 convert, zero pad
    int r = i >> 7, c = i & 127;
    lA[i] = (_Float16)((c < HDIM) ? state[sg[r] * HDIM + c] : 0.0f);
  }
  __syncthreads();
  for (int t = wave; t < 12; t += 4) {              // layer1: 48x128 * 128x64, relu
    int m0 = (t >> 2) * 16, n0 = (t & 3) * 16;
    v8f c = mma_tile<KPAD>(lA, KPAD, m0, w1bt, KPAD, n0, lane);
    int col = n0 + (lane & 15), rb = m0 + ((lane >> 4) << 3);
    float bv = (col < HID) ? b1[col] : 0.0f;
#pragma unroll
    for (int r = 0; r < 8; ++r) {
      float v = c[r] + bv;
      lH[(rb + r) * NPAD1 + col] = (_Float16)(v > 0.0f ? v : 0.0f);
    }
  }
  __syncthreads();
  for (int t = wave; t < 21; t += 4) {              // layer2: 48x64 * 64x112
    int m0 = (t / 7) * 16, n0 = (t % 7) * 16;
    v8f c = mma_tile<NPAD1>(lH, NPAD1, m0, w2bt, NPAD1, n0, lane);
    int col = n0 + (lane & 15), rb = m0 + ((lane >> 4) << 3);
    float bv = (col < HDIM) ? b2[col] : 0.0f;
#pragma unroll
    for (int r = 0; r < 8; ++r) lP[(rb + r) * NPAD2 + col] = c[r] + bv;
  }
  __syncthreads();
  if (sidx == nullptr) {                            // fwd: dst sorted, sum triples
    int rowbase = blockIdx.x * 16;
    for (int i = tid; i < 16 * HDIM; i += 128) {
      int r = i / HDIM, j = i - r * HDIM;
      agg[(rowbase + r) * NPAD2 + j] = lP[(3*r) * NPAD2 + j] +
                                       lP[(3*r+1) * NPAD2 + j] +
                                       lP[(3*r+2) * NPAD2 + j];
    }
  } else {                                          // bwd: random scatter via atomics
    for (int i = tid; i < 48 * HDIM; i += 128) {
      int e = i / HDIM, j = i - e * HDIM;
      atomicAdd(&agg[ss[e] * NPAD2 + j], lP[e * NPAD2 + j]);
    }
  }
}

// ---------------- node post-MLP + GRU step ----------------
__global__ void __launch_bounds__(128) node_gru_kernel(
    const float* __restrict__ agg, const float* __restrict__ xr, int node_base,
    const _Float16* __restrict__ w1bt, const float* __restrict__ b1,
    const _Float16* __restrict__ w2bt, const float* __restrict__ b2,
    const _Float16* __restrict__ whbt, const float* __restrict__ wi,
    const float* __restrict__ bi, const float* __restrict__ bh,
    float* __restrict__ state) {
  __shared__ _Float16 lAgg[16 * KPAD];
  __shared__ _Float16 lH2[16 * NPAD1];
  __shared__ float    lMsg[16 * NPAD2];
  __shared__ _Float16 lMsgH[16 * KPAD];
  __shared__ float    lGh[16 * GHPAD];
  __shared__ float    lX[16 * FDIM];
  const int tid = threadIdx.x, lane = tid & 31, wave = tid >> 5;
  const int row0 = blockIdx.x * 16;
  for (int i = tid; i < 16 * KPAD; i += 128) {
    int r = i >> 7, c = i & 127;
    lAgg[i]  = (_Float16)((c < HDIM) ? agg[(row0 + r) * NPAD2 + c] : 0.0f);
    lMsgH[i] = (_Float16)0.0f;
  }
  if (tid < 16 * FDIM) {
    int r = tid >> 2, f = tid & 3;
    lX[tid] = xr[(node_base + row0 + r) * FDIM + f];
  }
  __syncthreads();
  {                                                 // post layer1 (4 tiles, 1/wave)
    int n0 = wave * 16;
    v8f c = mma_tile<KPAD>(lAgg, KPAD, 0, w1bt, KPAD, n0, lane);
    int col = n0 + (lane & 15), rb = (lane >> 4) << 3;
    float bv = (col < HID) ? b1[col] : 0.0f;
#pragma unroll
    for (int r = 0; r < 8; ++r) {
      float v = c[r] + bv;
      lH2[(rb + r) * NPAD1 + col] = (_Float16)(v > 0.0f ? v : 0.0f);
    }
  }
  __syncthreads();
  for (int t = wave; t < 7; t += 4) {               // post layer2 -> msg
    int n0 = t * 16;
    v8f c = mma_tile<NPAD1>(lH2, NPAD1, 0, w2bt, NPAD1, n0, lane);
    int col = n0 + (lane & 15), rb = (lane >> 4) << 3;
    float bv = (col < HDIM) ? b2[col] : 0.0f;
#pragma unroll
    for (int r = 0; r < 8; ++r) {
      float v = c[r] + bv;
      lMsg[(rb + r) * NPAD2 + col] = v;
      if (col < HDIM) lMsgH[(rb + r) * KPAD + col] = (_Float16)v;
    }
  }
  __syncthreads();
  for (int t = wave; t < 19; t += 4) {              // gh = msg @ wh^T + bh
    int n0 = t * 16;
    v8f c = mma_tile<KPAD>(lMsgH, KPAD, 0, whbt, KPAD, n0, lane);
    int col = n0 + (lane & 15), rb = (lane >> 4) << 3;
    float bv = (col < 3 * HDIM) ? bh[col] : 0.0f;
#pragma unroll
    for (int r = 0; r < 8; ++r) lGh[(rb + r) * GHPAD + col] = c[r] + bv;
  }
  __syncthreads();
  for (int i = tid; i < 16 * HDIM; i += 128) {      // GRU gates (VALU)
    int r = i / HDIM, j = i - r * HDIM;
    float x0 = lX[r*4], x1 = lX[r*4+1], x2 = lX[r*4+2], x3 = lX[r*4+3];
    const float* w = wi + j * 4;
    float gr = bi[j]        + x0*w[0] + x1*w[1] + x2*w[2] + x3*w[3];
    w = wi + (HDIM + j) * 4;
    float gz = bi[HDIM+j]   + x0*w[0] + x1*w[1] + x2*w[2] + x3*w[3];
    w = wi + (2*HDIM + j) * 4;
    float gn = bi[2*HDIM+j] + x0*w[0] + x1*w[1] + x2*w[2] + x3*w[3];
    float hr = lGh[r*GHPAD + j];
    float hz = lGh[r*GHPAD + HDIM + j];
    float hn = lGh[r*GHPAD + 2*HDIM + j];
    float rg = 1.0f / (1.0f + __expf(-(gr + hr)));
    float zg = 1.0f / (1.0f + __expf(-(gz + hz)));
    float ng = tanhf(gn + rg * hn);
    state[(node_base + row0 + r) * HDIM + j] = (1.0f - zg) * ng + zg * lMsg[r*NPAD2 + j];
  }
}

// ---------------- projection xr = state @ proj_w + proj_b ----------------
__global__ void __launch_bounds__(128) proj_kernel(const float* __restrict__ state,
                                                   const float* __restrict__ pw,
                                                   const float* __restrict__ pb,
                                                   float* __restrict__ xr) {
  int node = blockIdx.x * 128 + threadIdx.x;
  if (node >= NTOT) return;
  float a0 = pb[0], a1 = pb[1], a2 = pb[2], a3 = pb[3];
  const float* s = state + node * HDIM;
  for (int j = 0; j < HDIM; ++j) {
    float v = s[j];
    const float* w = pw + j * 4;
    a0 += v * w[0]; a1 += v * w[1]; a2 += v * w[2]; a3 += v * w[3];
  }
  float* o = xr + node * 4;
  o[0] = a0; o[1] = a1; o[2] = a2; o[3] = a3;
}

// ---------------- classifier: sigmoid(relu(s@w1+b1)@w2+b2) * lit_mask ----------------
__global__ void __launch_bounds__(64) cls_kernel(const float* __restrict__ state,
    const _Float16* __restrict__ w1bt, const float* __restrict__ b1,
    const float* __restrict__ w2, const float* __restrict__ b2,
    float* __restrict__ pred) {
  __shared__ _Float16 lS[16 * KPAD];
  __shared__ float    lHid[16 * CLSP];
  const int tid = threadIdx.x, lane = tid & 31, wave = tid >> 5;
  const int row0 = blockIdx.x * 16;
  for (int i = tid; i < 16 * KPAD; i += 64) {
    int r = i >> 7, c = i & 127;
    lS[i] = (_Float16)((c < HDIM) ? state[(row0 + r) * HDIM + c] : 0.0f);
  }
  __syncthreads();
  {
    int n0 = wave * 16;
    v8f c = mma_tile<KPAD>(lS, KPAD, 0, w1bt, KPAD, n0, lane);
    int col = n0 + (lane & 15), rb = (lane >> 4) << 3;
    float bv = (col < DMID) ? b1[col] : 0.0f;
#pragma unroll
    for (int r = 0; r < 8; ++r) {
      float v = c[r] + bv;
      lHid[(rb + r) * CLSP + col] = v > 0.0f ? v : 0.0f;
    }
  }
  __syncthreads();
  if (tid < 16) {
    float acc = b2[0];
#pragma unroll
    for (int m = 0; m < DMID; ++m) acc += lHid[tid * CLSP + m] * w2[m];
    float p = 1.0f / (1.0f + __expf(-acc));
    int node = row0 + tid;
    pred[node] = (node < NPL) ? p : 0.0f;
  }
}

// ---------------- hard gate evaluation (per level, sequential) ----------------
__global__ void __launch_bounds__(128) gate_eval_kernel(float* __restrict__ pred,
    const int* __restrict__ src, const float* __restrict__ x, int level) {
  int i = blockIdx.x * 128 + threadIdx.x;
  if (i >= NPL) return;
  int d = level * NPL + i;
  float xj0 = pred[src[3*i]], xj1 = pred[src[3*i+1]], xj2 = pred[src[3*i+2]];
  float a0 = xj0 * HARD_INV_T, a1 = xj1 * HARD_INV_T, a2 = xj2 * HARD_INV_T;
  float mx = fmaxf(a0, fmaxf(a1, a2));
  float e0 = __expf(a0-mx), e1 = __expf(a1-mx), e2 = __expf(a2-mx);
  float si = 1.0f / (e0 + e1 + e2);
  float mn = fminf(a0, fminf(a1, a2));
  float f0 = __expf(mn-a0), f1 = __expf(mn-a1), f2 = __expf(mn-a2);
  float sj = 1.0f / (f0 + f1 + f2);
  float andm = (x[d*4+1] == 1.0f) ? 1.0f : 0.0f;
  float orm  = (x[d*4+2] == 1.0f) ? 1.0f : 0.0f;
  float notm = (x[d*4+3] == 1.0f) ? 1.0f : 0.0f;
  float smax = (e0*xj0 + e1*xj1 + e2*xj2) * si;
  float smin = (f0*xj0 + f1*xj1 + f2*xj2) * sj;
  pred[d] = andm * smin + orm * smax + notm * (3.0f - (xj0 + xj1 + xj2));
}

// ---------------- host orchestration ----------------
extern "C" void kernel_launch(void* const* d_in, const int* in_sizes, int n_in,
                              void* d_out, int out_size, void* d_ws, size_t ws_size,
                              hipStream_t stream) {
  (void)in_sizes; (void)n_in; (void)out_size; (void)ws_size;
  const float* x        = (const float*)d_in[0];
  const float* fpre_w1  = (const float*)d_in[1];
  const float* fpre_b1  = (const float*)d_in[2];
  const float* fpre_w2  = (const float*)d_in[3];
  const float* fpre_b2  = (const float*)d_in[4];
  const float* fpost_w1 = (const float*)d_in[5];
  const float* fpost_b1 = (const float*)d_in[6];
  const float* fpost_w2 = (const float*)d_in[7];
  const float* fpost_b2 = (const float*)d_in[8];
  const float* bpre_w1  = (const float*)d_in[9];
  const float* bpre_b1  = (const float*)d_in[10];
  const float* bpre_w2  = (const float*)d_in[11];
  const float* bpre_b2  = (const float*)d_in[12];
  const float* bpost_w1 = (const float*)d_in[13];
  const float* bpost_b1 = (const float*)d_in[14];
  const float* bpost_w2 = (const float*)d_in[15];
  const float* bpost_b2 = (const float*)d_in[16];
  const float* gru_f_wi = (const float*)d_in[17];
  const float* gru_f_wh = (const float*)d_in[18];
  const float* gru_f_bi = (const float*)d_in[19];
  const float* gru_f_bh = (const float*)d_in[20];
  const float* gru_b_wi = (const float*)d_in[21];
  const float* gru_b_wh = (const float*)d_in[22];
  const float* gru_b_bi = (const float*)d_in[23];
  const float* gru_b_bh = (const float*)d_in[24];
  const float* proj_w   = (const float*)d_in[25];
  const float* proj_b   = (const float*)d_in[26];
  const float* cls_w1   = (const float*)d_in[27];
  const float* cls_b1   = (const float*)d_in[28];
  const float* cls_w2   = (const float*)d_in[29];
  const float* cls_b2   = (const float*)d_in[30];
  const int*   src      = (const int*)d_in[31];
  const int*   dst      = (const int*)d_in[32];
  // n_levels=12, num_rounds=2: fixed by setup_inputs.

  char* ws = (char*)d_ws;
  size_t off = 0;
  auto alloc = [&](size_t bytes) -> char* {
    off = (off + 255) & ~(size_t)255;
    char* p = ws + off; off += bytes; return p;
  };
  float* state = (float*)alloc((size_t)NTOT * HDIM * 4);
  float* xr    = (float*)alloc((size_t)NTOT * FDIM * 4);
  float* agg   = (float*)alloc((size_t)NPL * NPAD2 * 4);
  float* pred  = (float*)alloc((size_t)NTOT * 4);
  _Float16* FPRE1  = (_Float16*)alloc(64 * 128 * 2);
  _Float16* FPRE2  = (_Float16*)alloc(112 * 64 * 2);
  _Float16* FPOST1 = (_Float16*)alloc(64 * 128 * 2);
  _Float16* FPOST2 = (_Float16*)alloc(112 * 64 * 2);
  _Float16* BPRE1  = (_Float16*)alloc(64 * 128 * 2);
  _Float16* BPRE2  = (_Float16*)alloc(112 * 64 * 2);
  _Float16* BPOST1 = (_Float16*)alloc(64 * 128 * 2);
  _Float16* BPOST2 = (_Float16*)alloc(112 * 64 * 2);
  _Float16* WHF    = (_Float16*)alloc(304 * 128 * 2);
  _Float16* WHB    = (_Float16*)alloc(304 * 128 * 2);
  _Float16* CLS1   = (_Float16*)alloc(32 * 128 * 2);

  hipMemsetAsync(state, 0, (size_t)NTOT * HDIM * 4, stream);

  auto prep = [&](const float* w, _Float16* bt, int K, int N, int Kp, int Np, int tr) {
    int tot = Kp * Np;
    prep_bt_kernel<<<(tot + 255) / 256, 256, 0, stream>>>(w, bt, K, N, Kp, Np, tr);
  };
  prep(fpre_w1,  FPRE1,  HDIM, HID,  KPAD, NPAD1, 1);
  prep(fpre_w2,  FPRE2,  HID,  HDIM, NPAD1, NPAD2, 1);
  prep(fpost_w1, FPOST1, HDIM, HID,  KPAD, NPAD1, 1);
  prep(fpost_w2, FPOST2, HID,  HDIM, NPAD1, NPAD2, 1);
  prep(bpre_w1,  BPRE1,  HDIM, HID,  KPAD, NPAD1, 1);
  prep(bpre_w2,  BPRE2,  HID,  HDIM, NPAD1, NPAD2, 1);
  prep(bpost_w1, BPOST1, HDIM, HID,  KPAD, NPAD1, 1);
  prep(bpost_w2, BPOST2, HID,  HDIM, NPAD1, NPAD2, 1);
  prep(gru_f_wh, WHF,    HDIM, 3 * HDIM, KPAD, GHPAD, 0);  // wh is already (wh^T)^T
  prep(gru_b_wh, WHB,    HDIM, 3 * HDIM, KPAD, GHPAD, 0);
  prep(cls_w1,   CLS1,   HDIM, DMID, KPAD, CLSP, 1);

  for (int r = 0; r < 2; ++r) {
    const float* xrp = x;
    if (r > 0) {
      proj_kernel<<<(NTOT + 127) / 128, 128, 0, stream>>>(state, proj_w, proj_b, xr);
      xrp = xr;
    }
    for (int l = 1; l < NLV; ++l) {                 // forward sweep
      edge_pre_kernel<<<EPL / 48, 128, 0, stream>>>(state, src + (l-1)*EPL, nullptr, 0,
          FPRE1, fpre_b1, FPRE2, fpre_b2, agg);
      node_gru_kernel<<<NPL / 16, 128, 0, stream>>>(agg, xrp, l * NPL,
          FPOST1, fpost_b1, FPOST2, fpost_b2, WHF, gru_f_wi, gru_f_bi, gru_f_bh, state);
    }
    for (int lb = 1; lb < NLV; ++lb) {              // backward sweep
      int f = NLV - 1 - lb;
      hipMemsetAsync(agg, 0, (size_t)NPL * NPAD2 * 4, stream);
      edge_pre_kernel<<<EPL / 48, 128, 0, stream>>>(state, dst + f*EPL, src + f*EPL,
          f * NPL, BPRE1, bpre_b1, BPRE2, bpre_b2, agg);
      node_gru_kernel<<<NPL / 16, 128, 0, stream>>>(agg, xrp, f * NPL,
          BPOST1, bpost_b1, BPOST2, bpost_b2, WHB, gru_b_wi, gru_b_bi, gru_b_bh, state);
    }
  }
  cls_kernel<<<NTOT / 16, 64, 0, stream>>>(state, CLS1, cls_b1, cls_w2, cls_b2, pred);
  for (int l = 1; l < NLV; ++l)
    gate_eval_kernel<<<(NPL + 127) / 128, 128, 0, stream>>>(pred, src + (l-1)*EPL, x, l);
  hipMemcpyAsync(d_out, pred + (size_t)(NLV - 1) * NPL, (size_t)NPL * sizeof(float),
                 hipMemcpyDeviceToDevice, stream);
}
